// DLRM_Net_29437705847015
// MI455X (gfx1250) — compile-verified, matrix operand order checked by reading, not run on hardware
//
#include <hip/hip_runtime.h>
#include <hip/hip_bf16.h>
#include <math.h>

// ---------------------------------------------------------------------------
// DLRM inference for MI455X (gfx1250, wave32, WMMA).
//   B=4096, L=4, NT=26, V=100000, D=64
//   bot MLP: 13->512->256->64 (relu)   top MLP: 415->512->256->1 (sigmoid)
// Strategy: bf16 WMMA (16x16x32) for all big GEMMs (each wave owns a 32x64
// output strip: 8 accumulators, B fragments reused 2x, A fragments 4x),
// VALU for the K=13 layer and the N=1 head, LDS-staged gather+interaction
// kernel for the bandwidth-bound embedding path.
// ---------------------------------------------------------------------------

typedef __bf16 bf16_t;
typedef __attribute__((ext_vector_type(16))) __bf16 v16bf;
typedef __attribute__((ext_vector_type(8)))  __bf16 v8bf;
typedef __attribute__((ext_vector_type(8)))  float  v8f;

union BFPack { v16bf v; v8bf h[2]; };

#define BATCH 4096
#define NTBL  26
#define VOCAB 100000
#define EMBD  64

// ---------------------------------------------------------------------------
// Weight pack: fp32 [N,K] -> bf16 [N,Kp] with zero padding (Kp >= K).
// ---------------------------------------------------------------------------
__global__ void pack_bf16_kernel(const float* __restrict__ src,
                                 bf16_t* __restrict__ dst,
                                 int N, int K, int Kp) {
    int idx = blockIdx.x * blockDim.x + threadIdx.x;
    if (idx >= N * Kp) return;
    int n = idx / Kp;
    int k = idx - n * Kp;
    dst[idx] = (k < K) ? (bf16_t)src[n * K + k] : (bf16_t)0.0f;
}

// ---------------------------------------------------------------------------
// Bottom layer 0: h0 = relu(x @ bw0^T + bb0), K=13 (too small for WMMA).
// ---------------------------------------------------------------------------
__global__ void bot0_kernel(const float* __restrict__ x,     // [B,13]
                            const float* __restrict__ w,     // [512,13]
                            const float* __restrict__ bias,  // [512]
                            bf16_t* __restrict__ out) {      // [B,512] bf16
    int idx = blockIdx.x * blockDim.x + threadIdx.x;
    if (idx >= BATCH * 512) return;
    int m = idx >> 9;
    int n = idx & 511;
    const float* xr = x + m * 13;
    const float* wr = w + n * 13;
    float s = bias[n];
#pragma unroll
    for (int k = 0; k < 13; ++k) s += xr[k] * wr[k];
    out[idx] = (bf16_t)fmaxf(s, 0.0f);
}

// ---------------------------------------------------------------------------
// Generic bf16 WMMA GEMM:  C[M,N] = act(A[M,K] @ W[N,K]^T + bias)
//   - A, W bf16 row-major; K mult. of 32, N mult. of 64, M mult. of 32
//   - each wave computes a 32(M) x 64(N) strip: 8 accumulators; every B
//     fragment feeds 2 WMMAs, every A fragment feeds 4 -> halves W traffic
//     and gives the scheduler 8 independent WMMAs per k-step to hide loads
//   - operand VGPR layout per CDNA5 ISA 16-bit A-matrix 16x32:
//       lanes 0-15  : K = {k0+0..7, k0+16..23}, row = lane
//       lanes 16-31 : K = {k0+8..15, k0+24..31}, row = lane-16
//     (B operand = rows of W, identical layout since B = W^T columns)
//   - outputs optionally stored as f32 and/or bf16 (null to skip)
// ---------------------------------------------------------------------------
__global__ void gemm_bf16_wmma_kernel(const bf16_t* __restrict__ A,
                                      const bf16_t* __restrict__ W,
                                      const float*  __restrict__ bias,
                                      float*  __restrict__ outF,
                                      bf16_t* __restrict__ outB,
                                      int M, int N, int K, int relu) {
    const int lane  = threadIdx.x & 31;
    const int wave  = threadIdx.x >> 5;
    const int wpb   = blockDim.x >> 5;
    const int tile  = blockIdx.x * wpb + wave;
    const int ntn   = N >> 6;                 // 64-wide N tiles
    const int tm    = (tile / ntn) << 5;      // 32-row M strip
    const int tn    = (tile % ntn) << 6;
    if (tm >= M) return;                      // wave-uniform exit (EXEC stays full)

    const int row  = lane & 15;
    const int koff = (lane >> 4) << 3;        // 0 or 8

    v8f acc[2][4] = {};
    const bf16_t* aptr0 = A + (size_t)(tm + row) * K + koff;
    const bf16_t* aptr1 = aptr0 + (size_t)16 * K;

    for (int k0 = 0; k0 < K; k0 += 32) {
        BFPack a0, a1, b[4];
        a0.h[0] = *(const v8bf*)(aptr0 + k0);
        a0.h[1] = *(const v8bf*)(aptr0 + k0 + 16);
        a1.h[0] = *(const v8bf*)(aptr1 + k0);
        a1.h[1] = *(const v8bf*)(aptr1 + k0 + 16);
#pragma unroll
        for (int j = 0; j < 4; ++j) {
            const bf16_t* wptr = W + (size_t)(tn + j * 16 + row) * K + k0 + koff;
            b[j].h[0] = *(const v8bf*)(wptr);
            b[j].h[1] = *(const v8bf*)(wptr + 16);
        }
#pragma unroll
        for (int j = 0; j < 4; ++j) {
            acc[0][j] = __builtin_amdgcn_wmma_f32_16x16x32_bf16(
                false, a0.v, false, b[j].v, (short)0, acc[0][j], false, false);
            acc[1][j] = __builtin_amdgcn_wmma_f32_16x16x32_bf16(
                false, a1.v, false, b[j].v, (short)0, acc[1][j], false, false);
        }
    }

    // C/D layout: lane l, reg r -> m = tile_m + r + 8*(l>=16), n = tile_n + (l&15)
    const int ncol = lane & 15;
#pragma unroll
    for (int h = 0; h < 2; ++h) {
        const int mbase = tm + h * 16 + ((lane >> 4) << 3);
#pragma unroll
        for (int j = 0; j < 4; ++j) {
            const int n = tn + j * 16 + ncol;
            const float bz = bias ? bias[n] : 0.0f;
#pragma unroll
            for (int r = 0; r < 8; ++r) {
                float v = acc[h][j][r] + bz;
                if (relu) v = fmaxf(v, 0.0f);
                const size_t o = (size_t)(mbase + r) * N + n;
                if (outF) outF[o] = v;
                if (outB) outB[o] = (bf16_t)v;
            }
        }
    }
}

// ---------------------------------------------------------------------------
// Embedding gather + pairwise interaction. One 128-thread block per sample.
//   T[0]    = h[b]                 (bottom-MLP output, f32)
//   T[1+t]  = sum_l emb[t][idx[t,b,l]]   (fully coalesced 256B row reads)
//   R[b]    = [ bf16(h), bf16(dot(T_i,T_j)) for tril(27,-1), 0 pad ] (416 cols)
// ---------------------------------------------------------------------------
__global__ void interact_kernel(const float* __restrict__ h,    // [B,64]
                                const int*   __restrict__ idx,  // [NT,B,L]
                                const float* __restrict__ emb,  // [NT,V,D]
                                bf16_t* __restrict__ R) {       // [B,416]
    __shared__ float T[27 * 64];
    const int b   = blockIdx.x;
    const int tid = threadIdx.x;
    const int d   = tid & 63;
    const int g   = tid >> 6;   // 0 or 1

    if (g == 0) T[d] = h[(size_t)b * 64 + d];
    for (int t = g; t < NTBL; t += 2) {
        const int*   ip = idx + ((size_t)t * BATCH + b) * 4;
        const float* tb = emb + (size_t)t * VOCAB * EMBD;
        float s = 0.0f;
#pragma unroll
        for (int l = 0; l < 4; ++l) s += tb[(size_t)ip[l] * EMBD + d];
        T[(t + 1) * 64 + d] = s;
    }
    __syncthreads();

    bf16_t* r = R + (size_t)b * 416;
    if (tid < 64) r[tid] = (bf16_t)T[tid];
    if (tid == 0) r[415] = (bf16_t)0.0f;

    for (int p = tid; p < 351; p += blockDim.x) {
        // invert p -> (i,j): pairs before row i = i*(i-1)/2, row-major tril order
        int i = (int)((1.0f + sqrtf(1.0f + 8.0f * (float)p)) * 0.5f);
        while (i * (i - 1) / 2 > p) --i;
        while ((i + 1) * i / 2 <= p) ++i;
        const int j = p - i * (i - 1) / 2;
        const float* Ti = T + i * 64;
        const float* Tj = T + j * 64;
        float s = 0.0f;
#pragma unroll 8
        for (int k = 0; k < 64; ++k) s += Ti[k] * Tj[k];
        r[64 + p] = (bf16_t)s;
    }
}

// ---------------------------------------------------------------------------
// Top layer 2: p = sigmoid(z @ tw2^T + tb2), N=1, K=256. One wave per row.
// ---------------------------------------------------------------------------
__global__ void top2_kernel(const float* __restrict__ z,     // [B,256]
                            const float* __restrict__ w,     // [256]
                            const float* __restrict__ bias,  // [1]
                            float* __restrict__ out) {       // [B]
    const int gw   = (blockIdx.x * blockDim.x + threadIdx.x) >> 5;
    const int lane = threadIdx.x & 31;
    if (gw >= BATCH) return;
    const float* zr = z + (size_t)gw * 256;
    float s = 0.0f;
#pragma unroll
    for (int k = lane; k < 256; k += 32) s += zr[k] * w[k];
#pragma unroll
    for (int off = 16; off > 0; off >>= 1) s += __shfl_down(s, off, 32);
    if (lane == 0) out[gw] = 1.0f / (1.0f + __expf(-(s + bias[0])));
}

// ---------------------------------------------------------------------------
// Launcher
// ---------------------------------------------------------------------------
extern "C" void kernel_launch(void* const* d_in, const int* in_sizes, int n_in,
                              void* d_out, int out_size, void* d_ws, size_t ws_size,
                              hipStream_t stream) {
    const float* x    = (const float*)d_in[0];
    const int*   lSi  = (const int*)  d_in[1];
    const float* emb  = (const float*)d_in[2];
    const float* bw0  = (const float*)d_in[3];
    const float* bb0  = (const float*)d_in[4];
    const float* bw1  = (const float*)d_in[5];
    const float* bb1  = (const float*)d_in[6];
    const float* bw2  = (const float*)d_in[7];
    const float* bb2  = (const float*)d_in[8];
    const float* tw0  = (const float*)d_in[9];
    const float* tb0  = (const float*)d_in[10];
    const float* tw1  = (const float*)d_in[11];
    const float* tb1  = (const float*)d_in[12];
    const float* tw2  = (const float*)d_in[13];
    const float* tb2  = (const float*)d_in[14];
    float* out = (float*)d_out;

    // workspace carve-up (256B aligned)
    char* ws = (char*)d_ws;
    size_t off = 0;
    auto carve = [&](size_t bytes) {
        char* p = ws + off;
        off = (off + bytes + 255) & ~(size_t)255;
        return p;
    };
    bf16_t* wb1  = (bf16_t*)carve((size_t)256 * 512 * 2);   // bw1 bf16
    bf16_t* wb2  = (bf16_t*)carve((size_t)64  * 256 * 2);   // bw2 bf16
    bf16_t* wt0p = (bf16_t*)carve((size_t)512 * 416 * 2);   // tw0 bf16 padded K 415->416
    bf16_t* wt1  = (bf16_t*)carve((size_t)256 * 512 * 2);   // tw1 bf16
    bf16_t* h0b  = (bf16_t*)carve((size_t)BATCH * 512 * 2); // bot0 out (bf16)
    bf16_t* h1b  = (bf16_t*)carve((size_t)BATCH * 256 * 2); // bot1 out (bf16)
    float*  h2f  = (float*) carve((size_t)BATCH * 64  * 4); // bot2 out (f32)
    bf16_t* Rb   = (bf16_t*)carve((size_t)BATCH * 416 * 2); // interaction out (bf16, padded)
    bf16_t* z0b  = (bf16_t*)carve((size_t)BATCH * 512 * 2); // top0 out (bf16)
    float*  z1f  = (float*) carve((size_t)BATCH * 256 * 4); // top1 out (f32)
    (void)ws_size; (void)in_sizes; (void)n_in; (void)out_size;

    // 1) weight packs (fp32 -> bf16, pad top0 K to 416)
    pack_bf16_kernel<<<(256 * 512 + 255) / 256, 256, 0, stream>>>(bw1, wb1, 256, 512, 512);
    pack_bf16_kernel<<<(64  * 256 + 255) / 256, 256, 0, stream>>>(bw2, wb2, 64, 256, 256);
    pack_bf16_kernel<<<(512 * 416 + 255) / 256, 256, 0, stream>>>(tw0, wt0p, 512, 415, 416);
    pack_bf16_kernel<<<(256 * 512 + 255) / 256, 256, 0, stream>>>(tw1, wt1, 256, 512, 512);

    // 2) bottom MLP
    bot0_kernel<<<(BATCH * 512 + 255) / 256, 256, 0, stream>>>(x, bw0, bb0, h0b);

    {   // bot1: 4096x256, K=512
        int tiles = (BATCH / 32) * (256 / 64);
        gemm_bf16_wmma_kernel<<<(tiles + 7) / 8, 256, 0, stream>>>(
            h0b, wb1, bb1, nullptr, h1b, BATCH, 256, 512, 1);
    }
    {   // bot2: 4096x64, K=256 -> f32 h for interaction
        int tiles = (BATCH / 32) * (64 / 64);
        gemm_bf16_wmma_kernel<<<(tiles + 7) / 8, 256, 0, stream>>>(
            h1b, wb2, bb2, h2f, nullptr, BATCH, 64, 256, 1);
    }

    // 3) embedding gather + interaction -> R [4096,416] bf16
    interact_kernel<<<BATCH, 128, 0, stream>>>(h2f, lSi, emb, Rb);

    // 4) top MLP
    {   // top0: 4096x512, K=416 (padded 415)
        int tiles = (BATCH / 32) * (512 / 64);
        gemm_bf16_wmma_kernel<<<(tiles + 7) / 8, 256, 0, stream>>>(
            Rb, wt0p, tb0, nullptr, z0b, BATCH, 512, 416, 1);
    }
    {   // top1: 4096x256, K=512 -> f32 for the head
        int tiles = (BATCH / 32) * (256 / 64);
        gemm_bf16_wmma_kernel<<<(tiles + 7) / 8, 256, 0, stream>>>(
            z0b, wt1, tb1, z1f, nullptr, BATCH, 256, 512, 1);
    }

    // 5) head: sigmoid dot-product, one wave per row
    top2_kernel<<<(BATCH * 32 + 255) / 256, 256, 0, stream>>>(z1f, tw2, tb2, out);
}
